// NKIMoELayer_24970939859026
// MI455X (gfx1250) — matrix operand order repaired and checked by reading, not compiled, not run on hardware
//
#include <hip/hip_runtime.h>
#include <hip/hip_bf16.h>
#include <math.h>

// ---- problem constants (fixed by the reference) ----
constexpr int Bb = 2, Ss = 2048, Hh = 2048, Ee = 16, Kk = 8, Ii = 1408;
constexpr int T = Bb * Ss;        // 4096 tokens
constexpr int TWO_I = 2 * Ii;     // 2816
constexpr int LDA = 40;           // padded LDS row stride (elements) = 80B (16B aligned, conflict-light)

typedef __bf16 bf16;
typedef __attribute__((ext_vector_type(16))) __bf16 bf16x16;
typedef __attribute__((ext_vector_type(8)))  __bf16 bf16x8;
typedef __attribute__((ext_vector_type(4)))  __bf16 bf16x4;
typedef __attribute__((ext_vector_type(8)))  float  f32x8;

#if __has_builtin(__builtin_amdgcn_tensor_load_to_lds)
#define HAVE_TDM 1
typedef __attribute__((ext_vector_type(4))) unsigned int u32x4;
typedef __attribute__((ext_vector_type(8))) int          i32x8;
typedef __attribute__((ext_vector_type(4))) int          i32x4;
#else
#define HAVE_TDM 0
#endif

// ------------------------------------------------------------------
// WMMA helper: CDNA5 v_wmma_f32_16x16x32_bf16 (wave32)
// ------------------------------------------------------------------
__device__ __forceinline__ f32x8 wmma_bf16(bf16x16 a, bf16x16 b, f32x8 c) {
  return __builtin_amdgcn_wmma_f32_16x16x32_bf16(
      false, a, false, b, (short)0, c, false, false);
}

// A fragment (16x32 bf16, ISA 7.12.2): lane l: m=l%16, h=l/16.
// elements 0..7  <-> K = 8h..8h+7 ; elements 8..15 <-> K = 16+8h..16+8h+7
__device__ __forceinline__ bf16x16 load_a_frag(const bf16* As, int row_base, int lane) {
  const int m = lane & 15, h = lane >> 4;
  const bf16* p = As + (row_base + m) * LDA;
  union { bf16x16 v; bf16x8 half[2]; } u;
  u.half[0] = *(const bf16x8*)(p + 8 * h);
  u.half[1] = *(const bf16x8*)(p + 16 + 8 * h);
  return u.v;
}

// B fragment (32x16 bf16): lane l: n=l%16, h=l/16; elements 0..7 <-> K=8h..8h+7 ; elements 8..15 <-> K=16+8h..16+8h+7.
__device__ __forceinline__ bf16x16 load_b_frag(const bf16* Bs, int col_base, int lane) {
  const int n = lane & 15, h = lane >> 4;
  const bf16* p = Bs + (col_base + n) * LDA;
  union { bf16x16 v; bf16x8 half[2]; } u;
  u.half[0] = *(const bf16x8*)(p + 8 * h);
  u.half[1] = *(const bf16x8*)(p + 16 + 8 * h);
  return u.v;
}

__device__ __forceinline__ float fast_silu(float g) {
  return g * __builtin_amdgcn_rcpf(1.f + __expf(-g));
}

// ------------------------------------------------------------------
// Kernel 1: router — logits, softmax, top-8 (descending), normalize
// ------------------------------------------------------------------
__global__ __launch_bounds__(256) void router_kernel(
    const float* __restrict__ x, const float* __restrict__ w,
    float* __restrict__ rw) {
  __shared__ float xs[Hh];
  __shared__ float logits[Ee];
  const int t = blockIdx.x;
  const float* xrow = x + (size_t)t * Hh;
  for (int i = threadIdx.x; i < Hh; i += 256) xs[i] = xrow[i];
  __syncthreads();
  if (threadIdx.x < Ee) {
    float acc = 0.f;
    for (int h = 0; h < Hh; ++h) acc += xs[h] * w[h * Ee + threadIdx.x];
    logits[threadIdx.x] = acc;
  }
  __syncthreads();
  if (threadIdx.x == 0) {
    float l[Ee];
    #pragma unroll
    for (int e = 0; e < Ee; ++e) l[e] = logits[e];
    float vals[Kk];
    #pragma unroll
    for (int j = 0; j < Kk; ++j) {          // selection: top-8 descending
      int bi = 0; float bv = -INFINITY;
      #pragma unroll
      for (int e = 0; e < Ee; ++e) if (l[e] > bv) { bv = l[e]; bi = e; }
      vals[j] = bv; l[bi] = -INFINITY;
    }
    const float m = vals[0];
    float ev[Kk], s = 0.f;
    #pragma unroll
    for (int j = 0; j < Kk; ++j) { ev[j] = __expf(vals[j] - m); s += ev[j]; }
    const float inv = __builtin_amdgcn_rcpf(s);
    #pragma unroll
    for (int j = 0; j < Kk; ++j) rw[(size_t)t * Kk + j] = ev[j] * inv;
  }
}

// ------------------------------------------------------------------
// Kernel 2: H_k[t][i] = w[t,k] * silu(X*Wg)*(X*Wu)  (bf16, into ws)
// block 128x128, 8 waves = 2(M)x4(N), wave tile 64x32 gate+up.
// grid = (I/128, T/128, 8)
// ------------------------------------------------------------------
__global__ __launch_bounds__(256) void moe_gemm1_kernel(
    const float* __restrict__ x,        // T x H
    const float* __restrict__ gate_up,  // E x H x 2I
    const float* __restrict__ rw,       // T x K
    bf16* __restrict__ hbuf) {          // K x T x I (bf16, pre-scaled)
  __shared__ bf16 As[128 * LDA];
  __shared__ bf16 Bg[128 * LDA];
  __shared__ bf16 Bu[128 * LDA];
  __shared__ float Wls[128];

  const int n0 = blockIdx.x * 128;
  const int m0 = blockIdx.y * 128;
  const int ex = blockIdx.z;
  const float* W = gate_up + (size_t)ex * Hh * TWO_I;

  const int tid = threadIdx.x;
  const int lane = tid & 31;
  const int wave = tid >> 5;
  const int wm = wave >> 2;  // 0..1 : 64-row block
  const int wn = wave & 3;   // 0..3 : 32-col block

  if (tid < 128) Wls[tid] = rw[(size_t)(m0 + tid) * Kk + ex];

  f32x8 accG[4][2] = {};
  f32x8 accU[4][2] = {};

  const int a_col4 = (tid & 7) * 4;   // A: f32 b128 loads along K
  const int a_row  = tid >> 3;        // 0..31 (+32*it)
  const int b_n    = tid & 127;       // B: dword loads along N, K-major LDS store
  const int b_kh   = tid >> 7;        // 0..1

  for (int kk0 = 0; kk0 < Hh; kk0 += 32) {
    __syncthreads();
    // ---- stage A tile: 128x32 f32 -> bf16, row-major ----
    #pragma unroll
    for (int it = 0; it < 4; ++it) {
      const int row = a_row + it * 32;
      const float4 v = *(const float4*)(x + (size_t)(m0 + row) * Hh + kk0 + a_col4);
      bf16x4 b; b[0] = (bf16)v.x; b[1] = (bf16)v.y; b[2] = (bf16)v.z; b[3] = (bf16)v.w;
      *(bf16x4*)(&As[row * LDA + a_col4]) = b;
    }
    // ---- stage B tiles (gate & up): 32(K) x 128(N), N-major/K-contig ----
    #pragma unroll
    for (int it = 0; it < 4; ++it) {
      const int kk4 = (b_kh + it * 2) * 4;
      bf16x4 g, u;
      #pragma unroll
      for (int j = 0; j < 4; ++j) {
        const size_t rowoff = (size_t)(kk0 + kk4 + j) * TWO_I;
        g[j] = (bf16)W[rowoff + n0 + b_n];
        u[j] = (bf16)W[rowoff + Ii + n0 + b_n];
      }
      *(bf16x4*)(&Bg[b_n * LDA + kk4]) = g;
      *(bf16x4*)(&Bu[b_n * LDA + kk4]) = u;
    }
    if (kk0 + 32 < Hh) {
      __builtin_prefetch(&W[(size_t)(kk0 + 32 + b_kh * 16) * TWO_I + n0 + b_n], 0, 1);
      __builtin_prefetch(&W[(size_t)(kk0 + 32 + b_kh * 16) * TWO_I + Ii + n0 + b_n], 0, 1);
    }
    __syncthreads();

    bf16x16 a[4], bg[2], bu[2];
    #pragma unroll
    for (int r = 0; r < 4; ++r) a[r] = load_a_frag(As, wm * 64 + r * 16, lane);
    #pragma unroll
    for (int c = 0; c < 2; ++c) {
      bg[c] = load_b_frag(Bg, wn * 32 + c * 16, lane);
      bu[c] = load_b_frag(Bu, wn * 32 + c * 16, lane);
    }
    #pragma unroll
    for (int r = 0; r < 4; ++r)
      #pragma unroll
      for (int c = 0; c < 2; ++c) {
        accG[r][c] = wmma_bf16(a[r], bg[c], accG[r][c]);
        accU[r][c] = wmma_bf16(a[r], bu[c], accU[r][c]);
      }
  }

  // ---- epilogue: w[t,k] * silu(g) * u, bf16 store ----
  const int ln = lane & 15, lh = lane >> 4;
  bf16* Hk = hbuf + (size_t)ex * T * Ii;
  #pragma unroll
  for (int r = 0; r < 4; ++r) {
    #pragma unroll
    for (int c = 0; c < 2; ++c) {
      const int rl0 = wm * 64 + r * 16 + lh * 8;
      const int col = n0 + wn * 32 + c * 16 + ln;
      const f32x8 g = accG[r][c], u = accU[r][c];
      #pragma unroll
      for (int v = 0; v < 8; ++v) {
        const float hid = fast_silu(g[v]) * u[v] * Wls[rl0 + v];
        Hk[(size_t)(m0 + rl0 + v) * Ii + col] = (bf16)hid;
      }
    }
  }
}

// ------------------------------------------------------------------
// Kernel 3: out = sum over (k,i) of Hs_k @ D_k — single merged GEMM,
// contraction dim K*I = 11264 (routing weight folded into hbuf).
// block 128x256, 8 waves = 2(M)x4(N), wave tile 64x64 -> 16 wmma/slab.
// H-tile staged via Tensor Data Mover when available.
// grid = (H/256, T/128)
// ------------------------------------------------------------------
__global__ __launch_bounds__(256) void moe_gemm2_kernel(
    const bf16*  __restrict__ hbuf,   // K x T x I (bf16, pre-scaled)
    const float* __restrict__ down,   // E x I x H
    float* __restrict__ out) {        // T x H
  __shared__ bf16 Hs[128 * LDA];
  __shared__ bf16 Ds[256 * LDA];

  const int n0 = blockIdx.x * 256;
  const int m0 = blockIdx.y * 128;
  const int tid = threadIdx.x;
  const int lane = tid & 31;
  const int wave = tid >> 5;
  const int wm = wave >> 2;  // 0..1
  const int wn = wave & 3;   // 0..3

  f32x8 acc[4][4] = {};

  const int h_col8 = (tid & 3) * 8;
  const int h_row  = tid >> 2;
  const int d_n    = tid;

#if HAVE_TDM
  const unsigned int lds_base = (unsigned int)(size_t)(&Hs[0]);
#endif

  for (int ex = 0; ex < Kk; ++ex) {
    const bf16*  Hk = hbuf + (size_t)ex * T * Ii;
    const float* Dk = down + (size_t)ex * Ii * Hh;
    for (int kk0 = 0; kk0 < Ii; kk0 += 32) {
      __syncthreads();
#if HAVE_TDM
      // ---- stage H tile via TDM: one wave-level DMA for 128x32 bf16,
      //      with HW LDS padding: 64B rows + 16B pad == LDA(40)*2B stride.
      if (wave == 0) {
        const unsigned long long ga =
            (unsigned long long)(const void*)(Hk + (size_t)m0 * Ii + kk0);
        u32x4 g0;
        g0[0] = 1u;                                    // count=1 (valid D#)
        g0[1] = lds_base;                              // lds_addr
        g0[2] = (unsigned int)ga;                      // global_addr[31:0]
        g0[3] = (unsigned int)((ga >> 32) & 0x01FFFFFFu) | (2u << 30);  // addr[56:32] | type=2
        i32x8 g1;
        g1[0] = (1 << 16)        // data_size = 1 -> 2 bytes
              | (1 << 20)        // pad_enable
              | (3 << 22)        // pad_interval: 16 DWORDs (=64B row)
              | (3 << 25);       // pad_amount:   4 DWORDs (=16B pad)
        const unsigned int td0 = 1u << 20, td1 = 1u << 20;  // large dims: no OOB clipping
        g1[1] = (int)((td0 & 0xFFFFu) << 16);
        g1[2] = (int)(((td0 >> 16) & 0xFFFFu) | ((td1 & 0xFFFFu) << 16));
        g1[3] = (int)(((td1 >> 16) & 0xFFFFu) | (32u << 16));  // tile_dim0 = 32
        g1[4] = 128;                                           // tile_dim1 = 128, tile_dim2 = 0
        g1[5] = Ii;                                            // tensor_dim0_stride (lo 32)
        g1[6] = 0;                                             // stride hi | dim1_stride lo
        g1[7] = 0;
        const i32x4 gz4 = {0, 0, 0, 0};
        const i32x8 gz8 = {0, 0, 0, 0, 0, 0, 0, 0};
        // 6-arg form (clang-23 / therock headers): g0, g1, g2, g3, extra group, cpol
        __builtin_amdgcn_tensor_load_to_lds(g0, g1, gz4, gz4, gz8, 0);
      }
#else
      #pragma unroll
      for (int it = 0; it < 2; ++it) {
        const int row = h_row + it * 64;
        const bf16x8 v = *(const bf16x8*)(Hk + (size_t)(m0 + row) * Ii + kk0 + h_col8);
        *(bf16x8*)(&Hs[row * LDA + h_col8]) = v;
      }
#endif
      // ---- stage D tile: 32(K) x 256(N), N-major/K-contig (f32 -> bf16) ----
      #pragma unroll
      for (int it = 0; it < 8; ++it) {
        const int kk4 = it * 4;
        bf16x4 d;
        #pragma unroll
        for (int j = 0; j < 4; ++j)
          d[j] = (bf16)Dk[(size_t)(kk0 + kk4 + j) * Hh + n0 + d_n];
        *(bf16x4*)(&Ds[d_n * LDA + kk4]) = d;
      }
      if (kk0 + 32 < Ii)
        __builtin_prefetch(&Dk[(size_t)(kk0 + 32) * Hh + n0 + d_n], 0, 1);
#if HAVE_TDM
      if (wave == 0) __builtin_amdgcn_s_wait_tensorcnt(0);
#endif
      __syncthreads();

      bf16x16 a[4], b[4];
      #pragma unroll
      for (int r = 0; r < 4; ++r) a[r] = load_a_frag(Hs, wm * 64 + r * 16, lane);
      #pragma unroll
      for (int c = 0; c < 4; ++c) b[c] = load_b_frag(Ds, wn * 64 + c * 16, lane);
      #pragma unroll
      for (int r = 0; r < 4; ++r)
        #pragma unroll
        for (int c = 0; c < 4; ++c)
          acc[r][c] = wmma_bf16(a[r], b[c], acc[r][c]);
    }
  }

  // ---- write out (f32) ----
  const int ln = lane & 15, lh = lane >> 4;
  #pragma unroll
  for (int r = 0; r < 4; ++r)
    #pragma unroll
    for (int c = 0; c < 4; ++c) {
      const int rowbase = m0 + wm * 64 + r * 16 + lh * 8;
      const int col = n0 + wn * 64 + c * 16 + ln;
      #pragma unroll
      for (int v = 0; v < 8; ++v)
        out[(size_t)(rowbase + v) * Hh + col] = acc[r][c][v];
    }
}

// ------------------------------------------------------------------
extern "C" void kernel_launch(void* const* d_in, const int* in_sizes, int n_in,
                              void* d_out, int out_size, void* d_ws, size_t ws_size,
                              hipStream_t stream) {
  const float* x       = (const float*)d_in[0];  // (B,S,H) f32
  const float* rw_w    = (const float*)d_in[1];  // (H,E)   f32
  const float* gate_up = (const float*)d_in[2];  // (E,H,2I) f32
  const float* down    = (const float*)d_in[3];  // (E,I,H)  f32

  float* out = (float*)d_out;                    // T*H floats
  float* rw  = out + (size_t)T * Hh;             // then T*K routing weights
  bf16*  hbuf = (bf16*)d_ws;                     // K*T*I bf16 (~92 MB scratch)

  router_kernel<<<T, 256, 0, stream>>>(x, rw_w, rw);

  dim3 g1(Ii / 128, T / 128, Kk);
  moe_gemm1_kernel<<<g1, 256, 0, stream>>>(x, gate_up, rw, hbuf);

  dim3 g2(Hh / 256, T / 128);
  moe_gemm2_kernel<<<g2, 256, 0, stream>>>(hbuf, down, out);
}
